// Injector_7945689497810
// MI455X (gfx1250) — matrix operand order repaired and checked by reading, not compiled, not run on hardware
//
#include <hip/hip_runtime.h>
#include <hip/hip_bf16.h>

#define NHh   12
#define NLl   3
#define NPp   4
#define DHd   32
#define DVv   384
#define EMB   768
#define BB    16
#define LQq   1024
#define LKVv  5376

typedef _Float16 f16;
typedef _Float16 v16h __attribute__((ext_vector_type(16)));
typedef _Float16 v8h  __attribute__((ext_vector_type(8)));
typedef float    v8f  __attribute__((ext_vector_type(8)));

// ---------------------------------------------------------------------------
// LayerNorm (fp32 in) -> f16 out, one block (256 thr) per row of length D
// ---------------------------------------------------------------------------
__global__ void ln_to_f16(const float* __restrict__ x, const float* __restrict__ w,
                          const float* __restrict__ b, f16* __restrict__ y, int D) {
  const int row = blockIdx.x;
  const int tid = threadIdx.x;
  const float* xr = x + (size_t)row * D;
  float s = 0.f, s2 = 0.f;
  for (int i = tid; i < D; i += 256) { float v = xr[i]; s += v; s2 += v * v; }
  __shared__ float sh[512];
  sh[tid] = s; sh[256 + tid] = s2;
  __syncthreads();
  for (int o = 128; o > 0; o >>= 1) {
    if (tid < o) { sh[tid] += sh[tid + o]; sh[256 + tid] += sh[256 + tid + o]; }
    __syncthreads();
  }
  const float mean = sh[0] / (float)D;
  const float var  = sh[256] / (float)D - mean * mean;
  const float rstd = rsqrtf(var + 1e-6f);
  f16* yr = y + (size_t)row * D;
  for (int i = tid; i < D; i += 256)
    yr[i] = (f16)((xr[i] - mean) * rstd * w[i] + b[i]);
}

// ---------------------------------------------------------------------------
// Transpose + fp32->f16 convert: in (R,C) row-major -> out (C,R) row-major
// ---------------------------------------------------------------------------
__global__ void transpose_cvt_f16(const float* __restrict__ in, f16* __restrict__ out,
                                  int R, int C) {
  int idx = blockIdx.x * blockDim.x + threadIdx.x;
  if (idx >= R * C) return;
  int r = idx / C, c = idx % C;
  out[(size_t)c * R + r] = (f16)in[idx];
}

// ---------------------------------------------------------------------------
// Register-blocked wave GEMM: one wave computes a (16*TM) x (16*TN) macro-tile.
// Per K-step: TM A-frag loads + TN B-frag loads feed TM*TN WMMAs.
// A: (M,K) f16 row-major; Bt: (N,K) f16 row-major (B transposed).
// Fragment layouts per CDNA5 ISA 7.12.2:
//   A frag  lane l: row m+(l&15); halves 0..7 = K k0+8*grp..+7,
//                   halves 8..15 = K k0+16+8*grp..+7
//   B frag  lane l: col n+(l&15); halves 0..7 = K k0+8*grp..+7,
//                   halves 8..15 = K k0+16+8*grp..+7
//   C/D     lane l, vgpr r: M = r + 8*grp, N = l&15
// ---------------------------------------------------------------------------
template <int TM, int TN>
__device__ __forceinline__ void wave_gemm_blk(const f16* __restrict__ A,
                                              const f16* __restrict__ Bt,
                                              int K, int m0, int n0, int lane,
                                              v8f acc[TM][TN]) {
  const int half = lane & 15;
  const int grp  = lane >> 4;
  const f16* arow[TM];
  const f16* brow[TN];
#pragma unroll
  for (int tm = 0; tm < TM; ++tm)
    arow[tm] = A + (size_t)(m0 + 16 * tm + half) * K;
#pragma unroll
  for (int tn = 0; tn < TN; ++tn)
    brow[tn] = Bt + (size_t)(n0 + 16 * tn + half) * K;

  for (int k0 = 0; k0 < K; k0 += 32) {
    v16h a[TM], b[TN];
#pragma unroll
    for (int tm = 0; tm < TM; ++tm) {
      v8h a0 = *(const v8h*)(arow[tm] + k0 + 8 * grp);
      v8h a1 = *(const v8h*)(arow[tm] + k0 + 16 + 8 * grp);
#pragma unroll
      for (int i = 0; i < 8; ++i) { a[tm][i] = a0[i]; a[tm][i + 8] = a1[i]; }
    }
#pragma unroll
    for (int tn = 0; tn < TN; ++tn) {
      v8h b0 = *(const v8h*)(brow[tn] + k0 + 8 * grp);
      v8h b1 = *(const v8h*)(brow[tn] + k0 + 16 + 8 * grp);
#pragma unroll
      for (int i = 0; i < 8; ++i) { b[tn][i] = b0[i]; b[tn][i + 8] = b1[i]; }
    }
#pragma unroll
    for (int tm = 0; tm < TM; ++tm)
#pragma unroll
      for (int tn = 0; tn < TN; ++tn)
        acc[tm][tn] = __builtin_amdgcn_wmma_f32_16x16x32_f16(
            false, a[tm], false, b[tn], (short)0, acc[tm][tn], false, false);
  }
}

// GEMM + bias -> f16 output (value projection). 64x32 macro-tile per wave.
__global__ void gemm_bias_f16out(const f16* __restrict__ A, const f16* __restrict__ Bt,
                                 const float* __restrict__ bias, f16* __restrict__ C,
                                 int M, int N, int K) {
  constexpr int TM = 4, TN = 2;
  const int tile = blockIdx.x * (blockDim.x >> 5) + (threadIdx.x >> 5);
  const int lane = threadIdx.x & 31;
  const int tilesN = N / (16 * TN);
  if (tile >= (M / (16 * TM)) * tilesN) return;
  const int m0 = (tile / tilesN) * 16 * TM;
  const int n0 = (tile % tilesN) * 16 * TN;
  v8f acc[TM][TN] = {};
  wave_gemm_blk<TM, TN>(A, Bt, K, m0, n0, lane, acc);
  const int half = lane & 15, grp = lane >> 4;
#pragma unroll
  for (int tn = 0; tn < TN; ++tn) {
    const int n = n0 + 16 * tn + half;
    const float bv = bias[n];
#pragma unroll
    for (int tm = 0; tm < TM; ++tm)
#pragma unroll
      for (int r = 0; r < 8; ++r) {
        const int m = m0 + 16 * tm + r + 8 * grp;
        C[(size_t)m * N + n] = (f16)(acc[tm][tn][r] + bv);
      }
  }
}

// GEMM + bias -> fp32 output (offsets, attention-weight logits). 64x16 per wave.
__global__ void gemm_bias_f32out(const f16* __restrict__ A, const f16* __restrict__ Bt,
                                 const float* __restrict__ bias, float* __restrict__ C,
                                 int M, int N, int K) {
  constexpr int TM = 4, TN = 1;
  const int tile = blockIdx.x * (blockDim.x >> 5) + (threadIdx.x >> 5);
  const int lane = threadIdx.x & 31;
  const int tilesN = N / 16;
  if (tile >= (M / (16 * TM)) * tilesN) return;
  const int m0 = (tile / tilesN) * 16 * TM;
  const int n0 = (tile % tilesN) * 16;
  v8f acc[TM][TN] = {};
  wave_gemm_blk<TM, TN>(A, Bt, K, m0, n0, lane, acc);
  const int half = lane & 15, grp = lane >> 4;
  const int n = n0 + half;
  const float bv = bias[n];
#pragma unroll
  for (int tm = 0; tm < TM; ++tm)
#pragma unroll
    for (int r = 0; r < 8; ++r) {
      const int m = m0 + 16 * tm + r + 8 * grp;
      C[(size_t)m * N + n] = acc[tm][0][r] + bv;
    }
}

// Final GEMM with fused residual epilogue: out = q + gamma * (A*B + b_out)
__global__ void gemm_out_epilogue(const f16* __restrict__ A, const f16* __restrict__ Bt,
                                  const float* __restrict__ bias, const float* __restrict__ q,
                                  const float* __restrict__ gamma, float* __restrict__ out,
                                  int M, int N, int K) {
  constexpr int TM = 4, TN = 2;
  const int tile = blockIdx.x * (blockDim.x >> 5) + (threadIdx.x >> 5);
  const int lane = threadIdx.x & 31;
  const int tilesN = N / (16 * TN);
  if (tile >= (M / (16 * TM)) * tilesN) return;
  const int m0 = (tile / tilesN) * 16 * TM;
  const int n0 = (tile % tilesN) * 16 * TN;
  v8f acc[TM][TN] = {};
  wave_gemm_blk<TM, TN>(A, Bt, K, m0, n0, lane, acc);
  const int half = lane & 15, grp = lane >> 4;
#pragma unroll
  for (int tn = 0; tn < TN; ++tn) {
    const int n = n0 + 16 * tn + half;
    const float bv = bias[n];
    const float g  = gamma[n];
#pragma unroll
    for (int tm = 0; tm < TM; ++tm)
#pragma unroll
      for (int r = 0; r < 8; ++r) {
        const int m = m0 + 16 * tm + r + 8 * grp;
        const size_t idx = (size_t)m * N + n;
        out[idx] = q[idx] + g * (acc[tm][tn][r] + bv);
      }
  }
}

// ---------------------------------------------------------------------------
// Deformable sampling: one wave per (b, q, h); lane = channel d (DH=32==wave32)
// softmax over NL*NP=12 logits, 4-corner zero-padded bilinear gather, f16 out.
// ---------------------------------------------------------------------------
__global__ void msda_sample(const f16* __restrict__ value, const float* __restrict__ off,
                            const float* __restrict__ awl, const float* __restrict__ ref,
                            const int* __restrict__ shapes, const int* __restrict__ lsi,
                            f16* __restrict__ attn) {
  const int wid  = blockIdx.x * (blockDim.x >> 5) + (threadIdx.x >> 5);
  const int lane = threadIdx.x & 31;
  if (wid >= BB * LQq * NHh) return;
  const int h  = wid % NHh;
  const int bq = wid / NHh;
  const int bI = bq / LQq;

  const float* aw12 = awl + (size_t)bq * (NHh * NLl * NPp) + h * (NLl * NPp);
  float w[NLl * NPp];
  float mx = -3.4e38f;
#pragma unroll
  for (int j = 0; j < NLl * NPp; ++j) { w[j] = aw12[j]; mx = fmaxf(mx, w[j]); }
  float ssum = 0.f;
#pragma unroll
  for (int j = 0; j < NLl * NPp; ++j) { w[j] = expf(w[j] - mx); ssum += w[j]; }
  const float inv = 1.0f / ssum;

  const float* offp = off + (size_t)bq * (NHh * NLl * NPp * 2) + h * (NLl * NPp * 2);
  const f16* vb = value + ((size_t)bI * LKVv) * DVv + h * DHd + lane;

  float acc = 0.0f;
  for (int l = 0; l < NLl; ++l) {
    const int H = shapes[2 * l], W = shapes[2 * l + 1], st = lsi[l];
    const float rx = ref[((size_t)bq * NLl + l) * 2 + 0];
    const float ry = ref[((size_t)bq * NLl + l) * 2 + 1];
#pragma unroll
    for (int p = 0; p < NPp; ++p) {
      const float lx = rx + offp[(l * NPp + p) * 2 + 0] / (float)W;
      const float ly = ry + offp[(l * NPp + p) * 2 + 1] / (float)H;
      const float x = lx * (float)W - 0.5f;
      const float y = ly * (float)H - 0.5f;
      const float fx = floorf(x), fy = floorf(y);
      const float dx = x - fx, dy = y - fy;
      const int x0 = (int)fx, y0 = (int)fy;
      const float wt = w[l * NPp + p] * inv;
      float s = 0.0f;
      if (x0 >= 0 && x0 < W && y0 >= 0 && y0 < H)
        s += (1.f - dx) * (1.f - dy) * (float)vb[(size_t)(st + y0 * W + x0) * DVv];
      if (x0 + 1 >= 0 && x0 + 1 < W && y0 >= 0 && y0 < H)
        s += dx * (1.f - dy) * (float)vb[(size_t)(st + y0 * W + x0 + 1) * DVv];
      if (x0 >= 0 && x0 < W && y0 + 1 >= 0 && y0 + 1 < H)
        s += (1.f - dx) * dy * (float)vb[(size_t)(st + (y0 + 1) * W + x0) * DVv];
      if (x0 + 1 >= 0 && x0 + 1 < W && y0 + 1 >= 0 && y0 + 1 < H)
        s += dx * dy * (float)vb[(size_t)(st + (y0 + 1) * W + x0 + 1) * DVv];
      acc += wt * s;
    }
  }
  attn[(size_t)bq * DVv + h * DHd + lane] = (f16)acc;
}

// ---------------------------------------------------------------------------
extern "C" void kernel_launch(void* const* d_in, const int* in_sizes, int n_in,
                              void* d_out, int out_size, void* d_ws, size_t ws_size,
                              hipStream_t stream) {
  (void)in_sizes; (void)n_in; (void)out_size; (void)ws_size;
  const float* q     = (const float*)d_in[0];
  const float* ref   = (const float*)d_in[1];
  const float* kv    = (const float*)d_in[2];
  const int*   shp   = (const int*)d_in[3];
  const int*   lsi   = (const int*)d_in[4];
  const float* ln1w  = (const float*)d_in[5];
  const float* ln1b  = (const float*)d_in[6];
  const float* ln2w  = (const float*)d_in[7];
  const float* ln2b  = (const float*)d_in[8];
  const float* gamma = (const float*)d_in[9];
  const float* w_off = (const float*)d_in[10];
  const float* b_off = (const float*)d_in[11];
  const float* w_aw  = (const float*)d_in[12];
  const float* b_aw  = (const float*)d_in[13];
  const float* w_val = (const float*)d_in[14];
  const float* b_val = (const float*)d_in[15];
  const float* w_out = (const float*)d_in[16];
  const float* b_out = (const float*)d_in[17];
  float* out = (float*)d_out;

  char* ws = (char*)d_ws;
  size_t o = 0;
  auto alloc = [&](size_t bytes) -> char* {
    char* p = ws + o;
    o += (bytes + 255) & ~(size_t)255;
    return p;
  };
  f16*   qn16    = (f16*)  alloc((size_t)BB * LQq  * EMB * 2);
  f16*   kvn16   = (f16*)  alloc((size_t)BB * LKVv * EMB * 2);
  f16*   wvalT   = (f16*)  alloc((size_t)EMB * DVv * 2);
  f16*   woffT   = (f16*)  alloc((size_t)EMB * (NHh * NLl * NPp * 2) * 2);
  f16*   wawT    = (f16*)  alloc((size_t)EMB * (NHh * NLl * NPp) * 2);
  f16*   woutT   = (f16*)  alloc((size_t)DVv * EMB * 2);
  f16*   value16 = (f16*)  alloc((size_t)BB * LKVv * DVv * 2);
  float* offb    = (float*)alloc((size_t)BB * LQq * (NHh * NLl * NPp * 2) * 4);
  float* awlb    = (float*)alloc((size_t)BB * LQq * (NHh * NLl * NPp) * 4);
  f16*   attn16  = (f16*)  alloc((size_t)BB * LQq * DVv * 2);

  // 1) LayerNorms -> f16
  ln_to_f16<<<BB * LQq,  256, 0, stream>>>(q,  ln1w, ln1b, qn16,  EMB);
  ln_to_f16<<<BB * LKVv, 256, 0, stream>>>(kv, ln2w, ln2b, kvn16, EMB);

  // 2) Weight transposes (row-major (K,N) fp32 -> (N,K) f16)
  auto tgrid = [](int n) { return (n + 255) / 256; };
  transpose_cvt_f16<<<tgrid(EMB * DVv), 256, 0, stream>>>(w_val, wvalT, EMB, DVv);
  transpose_cvt_f16<<<tgrid(EMB * 288), 256, 0, stream>>>(w_off, woffT, EMB, 288);
  transpose_cvt_f16<<<tgrid(EMB * 144), 256, 0, stream>>>(w_aw,  wawT,  EMB, 144);
  transpose_cvt_f16<<<tgrid(DVv * EMB), 256, 0, stream>>>(w_out, woutT, DVv, EMB);

  // 3) WMMA GEMMs (8 waves / block; 64x32 or 64x16 macro-tile per wave)
  auto ggrid = [](int M, int N, int tm, int tn) {
    return ((M / (16 * tm)) * (N / (16 * tn)) + 7) / 8;
  };
  gemm_bias_f16out<<<ggrid(BB * LKVv, DVv, 4, 2), 256, 0, stream>>>(
      kvn16, wvalT, b_val, value16, BB * LKVv, DVv, EMB);
  gemm_bias_f32out<<<ggrid(BB * LQq, 288, 4, 1), 256, 0, stream>>>(
      qn16, woffT, b_off, offb, BB * LQq, 288, EMB);
  gemm_bias_f32out<<<ggrid(BB * LQq, 144, 4, 1), 256, 0, stream>>>(
      qn16, wawT, b_aw, awlb, BB * LQq, 144, EMB);

  // 4) Deformable sampling: one wave per (b,q,h)
  msda_sample<<<(BB * LQq * NHh + 7) / 8, 256, 0, stream>>>(
      value16, offb, awlb, ref, shp, lsi, attn16);

  // 5) Output GEMM with fused residual epilogue
  gemm_out_epilogue<<<ggrid(BB * LQq, EMB, 4, 2), 256, 0, stream>>>(
      attn16, woutT, b_out, q, gamma, out, BB * LQq, EMB, DVv);
}
